// ConstituentEncoderLayer_88776974008778
// MI455X (gfx1250) — matrix operand order repaired and checked by reading, not compiled, not run on hardware
//
#include <hip/hip_runtime.h>

#define HEAD   12
#define DMODEL 768
#define DKV    64
#define DFF    3072
#define BSZ    8
#define SEQ    512
#define DK     64
#define NTOK   (BSZ*SEQ)      /* 4096 */
#define NBH    (BSZ*HEAD)     /* 96  */

typedef __attribute__((ext_vector_type(16))) __bf16 v16bf;
typedef __attribute__((ext_vector_type(8)))  float  v8f;

static __device__ __forceinline__ unsigned short f2bf(float f) {
  unsigned int u = __float_as_uint(f);
  u += 0x7FFFu + ((u >> 16) & 1u);            // round-to-nearest-even
  return (unsigned short)(u >> 16);
}
static __device__ __forceinline__ float bf2f(unsigned short h) {
  return __uint_as_float(((unsigned int)h) << 16);
}

// ---- WMMA fragment loaders (bf16, per ISA 7.12.2 layouts) -----------------
// A: 16x32 (MxK), row-major source, lda in elements.
static __device__ __forceinline__ v16bf load_frag_a(const unsigned short* __restrict__ A,
                                                    int lda, int m0, int k0, int lane) {
  int r  = m0 + (lane & 15);
  int hi = (lane >> 4) & 1;                    // lanes 16-31 hold K groups +8
  const unsigned short* p = A + (size_t)r * lda + k0 + hi * 8;
  union { uint4 u[2]; v16bf v; } U;
  U.u[0] = *(const uint4*)(p);                 // elems j=0..7  -> K = k0+hi*8+j
  U.u[1] = *(const uint4*)(p + 16);            // elems j=8..15 -> K = k0+16+hi*8+(j-8)
  return U.v;
}
// B: 32x16 (KxN), source stored [N][K] row-major (i.e. W[o][i]), ldb in elements.
static __device__ __forceinline__ v16bf load_frag_b(const unsigned short* __restrict__ B,
                                                    int ldb, int n0, int k0, int lane) {
  int n  = n0 + (lane & 15);
  int hi = (lane >> 4) & 1;                    // lanes 16-31 hold K groups +8
  const unsigned short* p = B + (size_t)n * ldb + k0 + hi * 8;
  union { uint4 u[2]; v16bf v; } U;
  U.u[0] = *(const uint4*)(p);
  U.u[1] = *(const uint4*)(p + 16);
  return U.v;
}

#define WMMA_BF16(a, b, c) \
  __builtin_amdgcn_wmma_f32_16x16x32_bf16(false, (a), false, (b), (short)0, (c), false, false)

// 2x2 register-blocked macro tile: 32M x 32N per wave, 4 WMMAs per K-step.
struct Acc2x2 { v8f c00, c01, c10, c11; };
static __device__ __forceinline__ Acc2x2 wmma_acc_2x2(const unsigned short* __restrict__ A, int lda,
                                                      const unsigned short* __restrict__ B, int ldb,
                                                      int m0, int n0, int K, int lane) {
  Acc2x2 ac;
  ac.c00 = (v8f){}; ac.c01 = (v8f){}; ac.c10 = (v8f){}; ac.c11 = (v8f){};
  for (int k0 = 0; k0 < K; k0 += 32) {
    v16bf a0 = load_frag_a(A, lda, m0,      k0, lane);
    v16bf a1 = load_frag_a(A, lda, m0 + 16, k0, lane);
    v16bf b0 = load_frag_b(B, ldb, n0,      k0, lane);
    v16bf b1 = load_frag_b(B, ldb, n0 + 16, k0, lane);
    ac.c00 = WMMA_BF16(a0, b0, ac.c00);
    ac.c01 = WMMA_BF16(a0, b1, ac.c01);
    ac.c10 = WMMA_BF16(a1, b0, ac.c10);
    ac.c11 = WMMA_BF16(a1, b1, ac.c11);
  }
  return ac;
}

// ---- elementwise / LN kernels ---------------------------------------------
__global__ void k_f32_to_bf16(const float* __restrict__ src, unsigned short* __restrict__ dst, int n) {
  int i = blockIdx.x * blockDim.x + threadIdx.x;
  int stride = gridDim.x * blockDim.x;
  for (; i < n; i += stride) dst[i] = f2bf(src[i]);
}

// One block (256 thr) per token; optionally produce two LN outputs (bf16).
__global__ void k_layernorm(const float* __restrict__ X,
                            const float* __restrict__ g0, const float* __restrict__ b0,
                            unsigned short* __restrict__ out0,
                            const float* __restrict__ g1, const float* __restrict__ b1,
                            unsigned short* __restrict__ out1) {
  __shared__ float red[256];
  __shared__ float sMean, sRstd;
  const int t = blockIdx.x, tid = threadIdx.x;
  const float* x = X + (size_t)t * DMODEL;
  float v0 = x[tid], v1 = x[tid + 256], v2 = x[tid + 512];
  red[tid] = v0 + v1 + v2;
  __syncthreads();
  for (int o = 128; o > 0; o >>= 1) { if (tid < o) red[tid] += red[tid + o]; __syncthreads(); }
  if (tid == 0) sMean = red[0] * (1.0f / DMODEL);
  __syncthreads();
  float mean = sMean;
  float d0 = v0 - mean, d1 = v1 - mean, d2 = v2 - mean;
  red[tid] = d0 * d0 + d1 * d1 + d2 * d2;
  __syncthreads();
  for (int o = 128; o > 0; o >>= 1) { if (tid < o) red[tid] += red[tid + o]; __syncthreads(); }
  if (tid == 0) sRstd = rsqrtf(red[0] * (1.0f / DMODEL) + 1e-5f);
  __syncthreads();
  float rstd = sRstd;
  size_t base = (size_t)t * DMODEL;
  float n0 = d0 * rstd, n1 = d1 * rstd, n2 = d2 * rstd;
  out0[base + tid]       = f2bf(n0 * g0[tid]       + b0[tid]);
  out0[base + tid + 256] = f2bf(n1 * g0[tid + 256] + b0[tid + 256]);
  out0[base + tid + 512] = f2bf(n2 * g0[tid + 512] + b0[tid + 512]);
  if (out1) {
    out1[base + tid]       = f2bf(n0 * g1[tid]       + b1[tid]);
    out1[base + tid + 256] = f2bf(n1 * g1[tid + 256] + b1[tid + 256]);
    out1[base + tid + 512] = f2bf(n2 * g1[tid + 512] + b1[tid + 512]);
  }
}

// ---- group attention: per-head q/k projection (shared 64x64 weights) ------
__global__ void k_ga_qk(const unsigned short* __restrict__ ctx,
                        const float* __restrict__ wq, const float* __restrict__ bq,
                        const float* __restrict__ wk, const float* __restrict__ bk,
                        float* __restrict__ qg, float* __restrict__ kg) {
  int idx = blockIdx.x * blockDim.x + threadIdx.x;           // NBH*SEQ*DK threads
  if (idx >= NBH * SEQ * DK) return;
  int d = idx & 63; int rest = idx >> 6; int s = rest & 511; int bh = rest >> 9;
  int b = bh / HEAD, h = bh % HEAD;
  const unsigned short* c = ctx + ((size_t)(b * SEQ + s)) * DMODEL + h * DK;
  const float* wqr = wq + d * DK;
  const float* wkr = wk + d * DK;
  float aq = 0.f, ak = 0.f;
  for (int i = 0; i < DK; i++) { float cv = bf2f(c[i]); aq += cv * wqr[i]; ak += cv * wkr[i]; }
  qg[idx] = aq + bq[d];
  kg[idx] = ak + bk[d];
}

// ---- group attention: neighbor scores, pair softmax, prefix scan ----------
__global__ void k_ga_scan(const float* __restrict__ qg, const float* __restrict__ kg,
                          const float* __restrict__ prior,
                          float* __restrict__ cumL, float* __restrict__ nsym) {
  __shared__ float nsupr[SEQ], nsubr[SEQ], Ls[SEQ], cum[SEQ];
  int bh = blockIdx.x, i = threadIdx.x;
  float p = prior[0];
  const float* qrow = qg + ((size_t)bh * SEQ + i) * DK;
  float es = -10000.0f, eb = -10000.0f;
  if (i < SEQ - 1) { const float* kr = kg + ((size_t)bh * SEQ + i + 1) * DK;
                     float a = 0.f; for (int t = 0; t < DK; t++) a += qrow[t] * kr[t]; es = a * (1.0f / DK); }
  if (i > 0)       { const float* kr = kg + ((size_t)bh * SEQ + i - 1) * DK;
                     float a = 0.f; for (int t = 0; t < DK; t++) a += qrow[t] * kr[t]; eb = a * (1.0f / DK); }
  float m = fmaxf(es, eb);
  float z0 = expf(es - m), z1 = expf(eb - m);
  float inv = 1.0f / (z0 + z1);
  nsupr[i] = z0 * inv;   // na[i, i+1]
  nsubr[i] = z1 * inv;   // na[i, i-1]
  __syncthreads();
  float ns_m = 0.f;
  if (i < SEQ - 1) {
    float v = sqrtf(nsupr[i] * nsubr[i + 1] + 1e-4f);     // symmetric na at [i,i+1]
    ns_m = p + (1.0f - p) * v;
    Ls[i] = logf(ns_m + 1e-9f);
  } else Ls[i] = 0.f;
  nsym[(size_t)bh * SEQ + i] = ns_m;
  __syncthreads();
  if (i == 0) { float a = 0.f; cum[0] = 0.f; for (int j = 1; j < SEQ; j++) { a += Ls[j - 1]; cum[j] = a; } }
  __syncthreads();
  cumL[(size_t)bh * SEQ + i] = cum[i];
}

// ---- bandwidth kernel: write group_prob & break_prob ----------------------
__global__ void k_ga_write(const float* __restrict__ cumL, const float* __restrict__ nsym,
                           const float* __restrict__ prior,
                           float* __restrict__ gp, float* __restrict__ bp) {
  int i = blockIdx.x, bh = blockIdx.y;
  float p = prior[0];
  float dval = p + (1.0f - p) * sqrtf(1e-4f);
  const float* cl = cumL + (size_t)bh * SEQ;
  const float* ns = nsym + (size_t)bh * SEQ;
  float ci = cl[i];
  size_t rowoff = ((size_t)bh * SEQ + i) * SEQ;
  for (int j = threadIdx.x; j < SEQ; j += blockDim.x) {
    float g;
    if (j == i) g = dval;
    else if (j > i) g = expf(cl[j] - ci) + 1e-4f;
    else            g = expf(ci - cl[j]) + 1e-4f;
    gp[rowoff + j] = g;
    int diff = j - i;
    float bv = dval;
    if (diff == 1)       bv = ns[i];
    else if (diff == -1) bv = ns[j];
    bp[rowoff + j] = bv;
  }
}

// ---- WMMA GEMMs (2x2 register-blocked: 32M x 32N per wave) -----------------
// QKV: xn0[4096x768] x {Wq,Wk,Wv}^T (+bias) -> q,k bf16 [bh][s][64], v bf16 transposed [bh][64][s]
static __device__ __forceinline__ void store_qkv_tile(v8f acc, int mt, int nl0, int which,
                                                      const float* __restrict__ bias,
                                                      unsigned short* __restrict__ q,
                                                      unsigned short* __restrict__ k,
                                                      unsigned short* __restrict__ vT, int lane) {
  int nl = nl0 + (lane & 15);
  int h = nl >> 6, d = nl & 63;
  float bsv = bias[nl];
  int mbase = mt + ((lane >> 4) << 3);
#pragma unroll
  for (int v = 0; v < 8; v++) {
    int t = mbase + v;
    int b_ = t >> 9, s = t & 511;
    unsigned short hv = f2bf(acc[v] + bsv);
    size_t bh = (size_t)(b_ * HEAD + h);
    if (which == 0)      q [(bh * SEQ + s) * DKV + d] = hv;
    else if (which == 1) k [(bh * SEQ + s) * DKV + d] = hv;
    else                 vT[(bh * DKV + d) * SEQ + s] = hv;
  }
}

__global__ void k_gemm_qkv(const unsigned short* __restrict__ xn,
                           const unsigned short* __restrict__ wqb, const unsigned short* __restrict__ wkb,
                           const unsigned short* __restrict__ wvb,
                           const float* __restrict__ bq, const float* __restrict__ bk, const float* __restrict__ bv,
                           unsigned short* __restrict__ q, unsigned short* __restrict__ k,
                           unsigned short* __restrict__ vT) {
  int lane = threadIdx.x & 31, wave = threadIdx.x >> 5;
  int m0 = blockIdx.x * 32;
  int n  = (blockIdx.y * 8 + wave) * 32;       // global output col macro, 0..2272 (72 macros)
  int which = n / DMODEL;                       // 32 | 768, macro never straddles
  int nloc0 = n - which * DMODEL;
  const unsigned short* B = which == 0 ? wqb : (which == 1 ? wkb : wvb);
  const float* bias = which == 0 ? bq : (which == 1 ? bk : bv);
  Acc2x2 ac = wmma_acc_2x2(xn, DMODEL, B, DMODEL, m0, nloc0, DMODEL, lane);
  store_qkv_tile(ac.c00, m0,      nloc0,      which, bias, q, k, vT, lane);
  store_qkv_tile(ac.c01, m0,      nloc0 + 16, which, bias, q, k, vT, lane);
  store_qkv_tile(ac.c10, m0 + 16, nloc0,      which, bias, q, k, vT, lane);
  store_qkv_tile(ac.c11, m0 + 16, nloc0 + 16, which, bias, q, k, vT, lane);
}

// scores = q k^T / 8  -> bf16 [bh][512][512]
static __device__ __forceinline__ void store_sc_tile(v8f acc, unsigned short* __restrict__ sc,
                                                     size_t base, int mt, int nt, int lane) {
  int n = nt + (lane & 15);
  int mbase = mt + ((lane >> 4) << 3);
#pragma unroll
  for (int v = 0; v < 8; v++) sc[base + (size_t)(mbase + v) * SEQ + n] = f2bf(acc[v] * 0.125f);
}

__global__ void k_gemm_scores(const unsigned short* __restrict__ q, const unsigned short* __restrict__ k,
                              unsigned short* __restrict__ sc) {
  int lane = threadIdx.x & 31, wave = threadIdx.x >> 5;
  int bh = blockIdx.z;
  int m0 = blockIdx.x * 32, n0 = (blockIdx.y * 8 + wave) * 32;
  const unsigned short* A = q + (size_t)bh * SEQ * DKV;
  const unsigned short* B = k + (size_t)bh * SEQ * DKV;   // [N=key][K=d] layout
  Acc2x2 ac = wmma_acc_2x2(A, DKV, B, DKV, m0, n0, DKV, lane);
  size_t base = (size_t)bh * SEQ * SEQ;
  store_sc_tile(ac.c00, sc, base, m0,      n0,      lane);
  store_sc_tile(ac.c01, sc, base, m0,      n0 + 16, lane);
  store_sc_tile(ac.c10, sc, base, m0 + 16, n0,      lane);
  store_sc_tile(ac.c11, sc, base, m0 + 16, n0 + 16, lane);
}

// in-place row softmax then multiply by group_prob; block=128, one row each
__global__ void k_softmax_mul(unsigned short* __restrict__ sc, const float* __restrict__ gp) {
  __shared__ float red[128];
  __shared__ float sMax, sSum;
  int r = blockIdx.x, bh = blockIdx.y, tid = threadIdx.x;
  unsigned short* row = sc + ((size_t)bh * SEQ + r) * SEQ;
  const float* gpr = gp + ((size_t)bh * SEQ + r) * SEQ;
  float x[4];
#pragma unroll
  for (int u = 0; u < 4; u++) x[u] = bf2f(row[tid + u * 128]);
  float mx = fmaxf(fmaxf(x[0], x[1]), fmaxf(x[2], x[3]));
  red[tid] = mx; __syncthreads();
  for (int o = 64; o > 0; o >>= 1) { if (tid < o) red[tid] = fmaxf(red[tid], red[tid + o]); __syncthreads(); }
  if (tid == 0) sMax = red[0];
  __syncthreads();
  mx = sMax;
  float e[4], s = 0.f;
#pragma unroll
  for (int u = 0; u < 4; u++) { e[u] = expf(x[u] - mx); s += e[u]; }
  red[tid] = s; __syncthreads();
  for (int o = 64; o > 0; o >>= 1) { if (tid < o) red[tid] += red[tid + o]; __syncthreads(); }
  if (tid == 0) sSum = red[0];
  __syncthreads();
  float inv = 1.0f / sSum;
#pragma unroll
  for (int u = 0; u < 4; u++) row[tid + u * 128] = f2bf(e[u] * inv * gpr[tid + u * 128]);
}

// out = att @ v ; fused residual: x1 = x + out  (block=64, 2 waves = 2 N-macros of 32)
static __device__ __forceinline__ void store_ao_tile(v8f acc, const float* __restrict__ xin,
                                                     float* __restrict__ xout,
                                                     int b_, int h, int mt, int nt, int lane) {
  int col = h * DKV + nt + (lane & 15);
  int mbase = mt + ((lane >> 4) << 3);
#pragma unroll
  for (int v = 0; v < 8; v++) {
    int s = mbase + v;
    size_t idx = ((size_t)(b_ * SEQ + s)) * DMODEL + col;
    xout[idx] = xin[idx] + acc[v];
  }
}

__global__ void k_gemm_attnout(const unsigned short* __restrict__ att, const unsigned short* __restrict__ vT,
                               const float* __restrict__ xin, float* __restrict__ xout) {
  int lane = threadIdx.x & 31, wave = threadIdx.x >> 5;
  int bh = blockIdx.y;
  int m0 = blockIdx.x * 32, n0 = wave * 32;
  const unsigned short* A = att + (size_t)bh * SEQ * SEQ;
  const unsigned short* B = vT + (size_t)bh * SEQ * DKV;   // [N=d][K=token]
  Acc2x2 ac = wmma_acc_2x2(A, SEQ, B, SEQ, m0, n0, SEQ, lane);
  int b_ = bh / HEAD, h = bh % HEAD;
  store_ao_tile(ac.c00, xin, xout, b_, h, m0,      n0,      lane);
  store_ao_tile(ac.c01, xin, xout, b_, h, m0,      n0 + 16, lane);
  store_ao_tile(ac.c10, xin, xout, b_, h, m0 + 16, n0,      lane);
  store_ao_tile(ac.c11, xin, xout, b_, h, m0 + 16, n0 + 16, lane);
}

// FFN1: h = relu(xn1 @ W1^T + b1) -> bf16 [4096][3072]
static __device__ __forceinline__ void store_ffn1_tile(v8f acc, const float* __restrict__ b1,
                                                       unsigned short* __restrict__ h,
                                                       int mt, int nt, int lane) {
  int n = nt + (lane & 15);
  float bb = b1[n];
  int mbase = mt + ((lane >> 4) << 3);
#pragma unroll
  for (int v = 0; v < 8; v++) h[(size_t)(mbase + v) * DFF + n] = f2bf(fmaxf(acc[v] + bb, 0.0f));
}

__global__ void k_gemm_ffn1(const unsigned short* __restrict__ xn1, const unsigned short* __restrict__ w1b,
                            const float* __restrict__ b1, unsigned short* __restrict__ h) {
  int lane = threadIdx.x & 31, wave = threadIdx.x >> 5;
  int m0 = blockIdx.x * 32, n0 = (blockIdx.y * 8 + wave) * 32;
  Acc2x2 ac = wmma_acc_2x2(xn1, DMODEL, w1b, DMODEL, m0, n0, DMODEL, lane);
  store_ffn1_tile(ac.c00, b1, h, m0,      n0,      lane);
  store_ffn1_tile(ac.c01, b1, h, m0,      n0 + 16, lane);
  store_ffn1_tile(ac.c10, b1, h, m0 + 16, n0,      lane);
  store_ffn1_tile(ac.c11, b1, h, m0 + 16, n0 + 16, lane);
}

// FFN2 fused residual: x += h @ W2^T + b2   (x lives in d_out)
static __device__ __forceinline__ void store_ffn2_tile(v8f acc, const float* __restrict__ b2,
                                                       float* __restrict__ x, int mt, int nt, int lane) {
  int n = nt + (lane & 15);
  float bb = b2[n];
  int mbase = mt + ((lane >> 4) << 3);
#pragma unroll
  for (int v = 0; v < 8; v++) {
    size_t idx = (size_t)(mbase + v) * DMODEL + n;
    x[idx] = x[idx] + acc[v] + bb;
  }
}

__global__ void k_gemm_ffn2(const unsigned short* __restrict__ h, const unsigned short* __restrict__ w2b,
                            const float* __restrict__ b2, float* __restrict__ x) {
  int lane = threadIdx.x & 31, wave = threadIdx.x >> 5;
  int m0 = blockIdx.x * 32, n0 = (blockIdx.y * 8 + wave) * 32;
  Acc2x2 ac = wmma_acc_2x2(h, DFF, w2b, DFF, m0, n0, DFF, lane);
  store_ffn2_tile(ac.c00, b2, x, m0,      n0,      lane);
  store_ffn2_tile(ac.c01, b2, x, m0,      n0 + 16, lane);
  store_ffn2_tile(ac.c10, b2, x, m0 + 16, n0,      lane);
  store_ffn2_tile(ac.c11, b2, x, m0 + 16, n0 + 16, lane);
}

// ---------------------------------------------------------------------------
extern "C" void kernel_launch(void* const* d_in, const int* in_sizes, int n_in,
                              void* d_out, int out_size, void* d_ws, size_t ws_size,
                              hipStream_t stream) {
  const float* x      = (const float*)d_in[0];
  const float* prior  = (const float*)d_in[2];
  const float* ga_g   = (const float*)d_in[3];
  const float* ga_b   = (const float*)d_in[4];
  const float* ga_wq  = (const float*)d_in[5];
  const float* ga_bq  = (const float*)d_in[6];
  const float* ga_wk  = (const float*)d_in[7];
  const float* ga_bk  = (const float*)d_in[8];
  const float* sa_wq  = (const float*)d_in[9];
  const float* sa_bq  = (const float*)d_in[10];
  const float* sa_wk  = (const float*)d_in[11];
  const float* sa_bk  = (const float*)d_in[12];
  const float* sa_wv  = (const float*)d_in[13];
  const float* sa_bv  = (const float*)d_in[14];
  const float* ff_w1  = (const float*)d_in[15];
  const float* ff_b1  = (const float*)d_in[16];
  const float* ff_w2  = (const float*)d_in[17];
  const float* ff_b2  = (const float*)d_in[18];
  const float* sl0_g  = (const float*)d_in[19];
  const float* sl0_b  = (const float*)d_in[20];
  const float* sl1_g  = (const float*)d_in[21];
  const float* sl1_b  = (const float*)d_in[22];

  float* xout = (float*)d_out;                                  // [8,512,768]
  float* gp   = xout + (size_t)NTOK * DMODEL;                   // [8,12,512,512]
  float* bp   = gp + (size_t)NBH * SEQ * SEQ;                   // [8,12,512,512]

  char* ws = (char*)d_ws;
  size_t off = 0;
  auto alloc = [&](size_t bytes) { size_t o = off; off = (off + bytes + 255) & ~(size_t)255; return o; };
  unsigned short* ctx  = (unsigned short*)(ws + alloc((size_t)NTOK * DMODEL * 2));
  unsigned short* xn0  = (unsigned short*)(ws + alloc((size_t)NTOK * DMODEL * 2));
  unsigned short* xn1  = (unsigned short*)(ws + alloc((size_t)NTOK * DMODEL * 2));
  unsigned short* wqb  = (unsigned short*)(ws + alloc((size_t)DMODEL * DMODEL * 2));
  unsigned short* wkb  = (unsigned short*)(ws + alloc((size_t)DMODEL * DMODEL * 2));
  unsigned short* wvb  = (unsigned short*)(ws + alloc((size_t)DMODEL * DMODEL * 2));
  unsigned short* w1b  = (unsigned short*)(ws + alloc((size_t)DFF * DMODEL * 2));
  unsigned short* w2b  = (unsigned short*)(ws + alloc((size_t)DMODEL * DFF * 2));
  unsigned short* qb   = (unsigned short*)(ws + alloc((size_t)NBH * SEQ * DKV * 2));
  unsigned short* kb   = (unsigned short*)(ws + alloc((size_t)NBH * SEQ * DKV * 2));
  unsigned short* vTb  = (unsigned short*)(ws + alloc((size_t)NBH * DKV * SEQ * 2));
  unsigned short* sc   = (unsigned short*)(ws + alloc((size_t)NBH * SEQ * SEQ * 2));
  unsigned short* hb   = (unsigned short*)(ws + alloc((size_t)NTOK * DFF * 2));
  float* qg   = (float*)(ws + alloc((size_t)NBH * SEQ * DK * 4));
  float* kg   = (float*)(ws + alloc((size_t)NBH * SEQ * DK * 4));
  float* cumL = (float*)(ws + alloc((size_t)NBH * SEQ * 4));
  float* nsym = (float*)(ws + alloc((size_t)NBH * SEQ * 4));

  // 1) weight conversions to bf16
  int nw = DMODEL * DMODEL;
  k_f32_to_bf16<<<(nw + 255) / 256, 256, 0, stream>>>(sa_wq, wqb, nw);
  k_f32_to_bf16<<<(nw + 255) / 256, 256, 0, stream>>>(sa_wk, wkb, nw);
  k_f32_to_bf16<<<(nw + 255) / 256, 256, 0, stream>>>(sa_wv, wvb, nw);
  int nf = DFF * DMODEL;
  k_f32_to_bf16<<<(nf + 255) / 256, 256, 0, stream>>>(ff_w1, w1b, nf);
  k_f32_to_bf16<<<(nf + 255) / 256, 256, 0, stream>>>(ff_w2, w2b, nf);

  // 2) dual layernorm: ctx (group-attn) and xn0 (sublayer0)
  k_layernorm<<<NTOK, 256, 0, stream>>>(x, ga_g, ga_b, ctx, sl0_g, sl0_b, xn0);

  // 3) group attention chain
  k_ga_qk<<<(NBH * SEQ * DK) / 256, 256, 0, stream>>>(ctx, ga_wq, ga_bq, ga_wk, ga_bk, qg, kg);
  k_ga_scan<<<NBH, SEQ, 0, stream>>>(qg, kg, prior, cumL, nsym);
  k_ga_write<<<dim3(SEQ, NBH), 256, 0, stream>>>(cumL, nsym, prior, gp, bp);

  // 4) self-attention (WMMA, 2x2 blocked)
  k_gemm_qkv<<<dim3(NTOK / 32, 9), 256, 0, stream>>>(xn0, wqb, wkb, wvb, sa_bq, sa_bk, sa_bv, qb, kb, vTb);
  k_gemm_scores<<<dim3(SEQ / 32, 2, NBH), 256, 0, stream>>>(qb, kb, sc);
  k_softmax_mul<<<dim3(SEQ, NBH), 128, 0, stream>>>(sc, gp);
  k_gemm_attnout<<<dim3(SEQ / 32, NBH), 64, 0, stream>>>(sc, vTb, x, xout);

  // 5) FFN (WMMA, 2x2 blocked) with residual
  k_layernorm<<<NTOK, 256, 0, stream>>>(xout, sl1_g, sl1_b, xn1, nullptr, nullptr, nullptr);
  k_gemm_ffn1<<<dim3(NTOK / 32, 12), 256, 0, stream>>>(xn1, w1b, ff_b1, hb);
  k_gemm_ffn2<<<dim3(NTOK / 32, 3), 256, 0, stream>>>(hb, w2b, ff_b2, xout);
}
